// EQI_79877801771589
// MI455X (gfx1250) — 8-bit matrix operand order repaired and checked by reading, compiled, not run on hardware
//
#include <hip/hip_runtime.h>
#include <stdint.h>

typedef __attribute__((ext_vector_type(8))) int v8i;
typedef __attribute__((ext_vector_type(4))) unsigned int u32x4;
typedef __attribute__((ext_vector_type(8))) int i32x8;
typedef __attribute__((ext_vector_type(4))) int i32x4;

#if __has_builtin(__builtin_amdgcn_tensor_load_to_lds) && \
    __has_builtin(__builtin_amdgcn_s_wait_tensorcnt)
#define EQI_HAVE_TDM 1
#else
#define EQI_HAVE_TDM 0
#endif

__global__ void pack_act_kernel(const int* __restrict__ src,
                                uint32_t* __restrict__ dst, int ndw) {
  int gid = blockIdx.x * blockDim.x + threadIdx.x;
  if (gid >= ndw) return;
  const int* s = src + (size_t)gid * 4;
  uint32_t w = ((uint32_t)(s[0] & 0xFF)) |
               ((uint32_t)(s[1] & 0xFF) << 8) |
               ((uint32_t)(s[2] & 0xFF) << 16) |
               ((uint32_t)(s[3] & 0xFF) << 24);
  dst[gid] = w;
}

__global__ void pack_w_kernel(const int* __restrict__ W, const int* __restrict__ E,
                              uint32_t* __restrict__ dst, int K, int N, int Npad) {
  int gid = blockIdx.x * blockDim.x + threadIdx.x;
  int kTiles = K >> 6;
  int total = (Npad >> 4) * kTiles * 256;
  if (gid >= total) return;
  int v    = gid & 7;
  int lane = (gid >> 3) & 31;
  int frag = gid >> 8;
  int kTile = frag % kTiles;
  int nTile = frag / kTiles;
  int n = nTile * 16 + (lane & 15);
  int h = lane >> 4;
  int kbase = kTile * 64 + ((v & 4) ? 32 : 0) + h * 16 + (v & 3) * 4;
  uint32_t w = 0;
  if (n < N) {
    #pragma unroll
    for (int b = 0; b < 4; ++b) {
      int k = kbase + b;
      int val = W[(size_t)k * N + n] + E[(size_t)k * N + n];
      w |= ((uint32_t)(val & 0xFF)) << (8 * b);
    }
  }
  dst[gid] = w;
}

template <int K, int NOUT, bool ADD_HID, bool WRITE_F32, bool WRITE_I8>
__global__ __launch_bounds__(256) void gemm_i8_kernel(
    const uint32_t* __restrict__ Ap,
    const uint32_t* __restrict__ Bp,
    const int* __restrict__ bw,
    const int* __restrict__ be,
    const int* __restrict__ hid,
    float* __restrict__ outF,
    uint8_t* __restrict__ outI8)
{
  constexpr int KT  = K / 64;
  constexpr int KT2 = K / 128;
  constexpr int LDW = 37;
  constexpr int ldA = K / 4;
  __shared__ uint32_t ldsA[2][128 * LDW];

  const int tid  = threadIdx.x;
  const int lane = tid & 31;
  const int wave = tid >> 5;
  const int wm   = wave & 1;
  const int wn   = wave >> 1;
  const int blockM = blockIdx.y * 128;
  const int blockN = blockIdx.x * 128;

  auto load_tile = [&](int kk, int buf) {
#if EQI_HAVE_TDM
    if (wave == 0) {
      uint64_t gaddr = (uint64_t)(uintptr_t)(Ap + (size_t)blockM * ldA + kk * 32);
      uint32_t laddr = (uint32_t)(uintptr_t)(&ldsA[buf][0]);
      u32x4 g0;
      g0[0] = 1u;
      g0[1] = laddr;
      g0[2] = (uint32_t)gaddr;
      g0[3] = ((uint32_t)(gaddr >> 32) & 0x01FFFFFFu) | (2u << 30);
      i32x8 g1;
      g1[0] = (2 << 16)
            | (1 << 20)
            | (4 << 22)
            | (4 << 25);
      g1[1] = (int)(((unsigned)ldA & 0xFFFFu) << 16);
      g1[2] = (int)((((unsigned)ldA >> 16) & 0xFFFFu) |
                    ((8192u & 0xFFFFu) << 16));
      g1[3] = (int)(((8192u >> 16) & 0xFFFFu) | (32u << 16));
      g1[4] = 128;
      g1[5] = ldA;
      g1[6] = 0;
      g1[7] = 0;
      i32x4 z4 = {0, 0, 0, 0};
#if defined(__clang_major__) && (__clang_major__ >= 23)
      i32x8 z8 = {0, 0, 0, 0, 0, 0, 0, 0};
      __builtin_amdgcn_tensor_load_to_lds(g0, g1, z4, z4, z8, 0);
#else
      __builtin_amdgcn_tensor_load_to_lds(g0, g1, z4, z4, 0);
#endif
    }
#else
    #pragma unroll
    for (int i = 0; i < 4; ++i) {
      int lin = tid + i * 256;
      int row = lin >> 3;
      int q   = lin & 7;
      uint4 d = *(const uint4*)(Ap + (size_t)(blockM + row) * ldA + kk * 32 + q * 4);
      uint32_t* p = &ldsA[buf][row * LDW + q * 4];
      p[0] = d.x; p[1] = d.y; p[2] = d.z; p[3] = d.w;
    }
#endif
  };

  const int m = lane & 15;
  const int h = lane >> 4;
  int kd[8];
  #pragma unroll
  for (int v = 0; v < 8; ++v)
    kd[v] = ((((v & 4) ? 32 : 0) + (v & 3) * 4 + h * 16) >> 2);

  v8i c00 = {}, c01 = {}, c10 = {}, c11 = {};
  v8i c20 = {}, c21 = {}, c30 = {}, c31 = {};

  const size_t bBase0 = ((size_t)((blockN >> 4) + wn * 2 + 0) * KT) * 256 + lane * 8;
  const size_t bBase1 = ((size_t)((blockN >> 4) + wn * 2 + 1) * KT) * 256 + lane * 8;

  load_tile(0, 0);
  for (int kk = 0; kk < KT2; ++kk) {
    const int buf = kk & 1;
#if EQI_HAVE_TDM
    if (wave == 0) __builtin_amdgcn_s_wait_tensorcnt(0);
#endif
    __syncthreads();
    if (kk + 1 < KT2) load_tile(kk + 1, buf ^ 1);

    const uint32_t* lp = &ldsA[buf][(wm * 64 + m) * LDW];
    #pragma unroll
    for (int s = 0; s < 2; ++s) {
      v8i b0 = *(const v8i*)(Bp + bBase0 + (size_t)(2 * kk + s) * 256);
      v8i b1 = *(const v8i*)(Bp + bBase1 + (size_t)(2 * kk + s) * 256);

      const int so = s * 16;
      v8i a0, a1, a2, a3;
      #pragma unroll
      for (int v = 0; v < 8; ++v) a0[v] = (int)lp[so + kd[v]];
      #pragma unroll
      for (int v = 0; v < 8; ++v) a1[v] = (int)lp[16 * LDW + so + kd[v]];
      #pragma unroll
      for (int v = 0; v < 8; ++v) a2[v] = (int)lp[32 * LDW + so + kd[v]];
      #pragma unroll
      for (int v = 0; v < 8; ++v) a3[v] = (int)lp[48 * LDW + so + kd[v]];

      c00 = __builtin_amdgcn_wmma_i32_16x16x64_iu8(true, a0, true, b0, c00, false, false);
      c01 = __builtin_amdgcn_wmma_i32_16x16x64_iu8(true, a0, true, b1, c01, false, false);
      c10 = __builtin_amdgcn_wmma_i32_16x16x64_iu8(true, a1, true, b0, c10, false, false);
      c11 = __builtin_amdgcn_wmma_i32_16x16x64_iu8(true, a1, true, b1, c11, false, false);
      c20 = __builtin_amdgcn_wmma_i32_16x16x64_iu8(true, a2, true, b0, c20, false, false);
      c21 = __builtin_amdgcn_wmma_i32_16x16x64_iu8(true, a2, true, b1, c21, false, false);
      c30 = __builtin_amdgcn_wmma_i32_16x16x64_iu8(true, a3, true, b0, c30, false, false);
      c31 = __builtin_amdgcn_wmma_i32_16x16x64_iu8(true, a3, true, b1, c31, false, false);
    }
  }

  v8i acc[4][2] = {{c00, c01}, {c10, c11}, {c20, c21}, {c30, c31}};
  #pragma unroll
  for (int ni = 0; ni < 2; ++ni) {
    int col = blockN + wn * 32 + ni * 16 + m;
    bool colOK = (col < NOUT);
    int bias = colOK ? (bw[col] + be[col]) : 0;
    #pragma unroll
    for (int mi = 0; mi < 4; ++mi) {
      #pragma unroll
      for (int v = 0; v < 8; ++v) {
        int row = blockM + wm * 64 + mi * 16 + h * 8 + v;
        int8_t val = (int8_t)(acc[mi][ni][v] + bias);
        if constexpr (ADD_HID)
          val = (int8_t)(val + (int8_t)hid[(size_t)row * 1024 + col]);
        if constexpr (WRITE_I8)
          outI8[(size_t)row * 1024 + col] = (uint8_t)val;
        if constexpr (WRITE_F32)
          if (colOK) outF[(size_t)row * NOUT + col] = (float)val;
      }
    }
  }
}

extern "C" void kernel_launch(void* const* d_in, const int* in_sizes, int n_in,
                              void* d_out, int out_size, void* d_ws, size_t ws_size,
                              hipStream_t stream) {
  (void)in_sizes; (void)n_in; (void)out_size; (void)ws_size;
  const int* W1  = (const int*)d_in[0];
  const int* b1  = (const int*)d_in[1];
  const int* W2  = (const int*)d_in[2];
  const int* b2  = (const int*)d_in[3];
  const int* W3  = (const int*)d_in[4];
  const int* b3  = (const int*)d_in[5];
  const int* e1  = (const int*)d_in[6];
  const int* eb1 = (const int*)d_in[7];
  const int* e2  = (const int*)d_in[8];
  const int* eb2 = (const int*)d_in[9];
  const int* e3  = (const int*)d_in[10];
  const int* eb3 = (const int*)d_in[11];
  const int* inputs  = (const int*)d_in[12];
  const int* hiddens = (const int*)d_in[13];

  float* outH2 = (float*)d_out;
  float* outO  = (float*)d_out + (size_t)8192 * 1024;

  uint8_t* ws = (uint8_t*)d_ws;
  size_t off = 0;
  auto walloc = [&](size_t bytes) -> uint8_t* {
    uint8_t* p = ws + off;
    off += (bytes + 255) & ~(size_t)255;
    return p;
  };
  uint32_t* A1p = (uint32_t*)walloc((size_t)8192 * 3072);
  uint32_t* A2p = (uint32_t*)walloc((size_t)8192 * 1024);
  uint32_t* A3p = (uint32_t*)walloc((size_t)8192 * 1024);
  uint32_t* W1p = (uint32_t*)walloc((size_t)3072 * 1024);
  uint32_t* W2p = (uint32_t*)walloc((size_t)1024 * 1024);
  uint32_t* W3p = (uint32_t*)walloc((size_t)1024 * 128);

  {
    int ndw = 8192 * 3072 / 4;
    pack_act_kernel<<<(ndw + 255) / 256, 256, 0, stream>>>(inputs, A1p, ndw);
  }
  {
    int tot = (1024 / 16) * (3072 / 64) * 256;
    pack_w_kernel<<<(tot + 255) / 256, 256, 0, stream>>>(W1, e1, W1p, 3072, 1024, 1024);
  }
  {
    int tot = (1024 / 16) * (1024 / 64) * 256;
    pack_w_kernel<<<(tot + 255) / 256, 256, 0, stream>>>(W2, e2, W2p, 1024, 1024, 1024);
  }
  {
    int tot = (128 / 16) * (1024 / 64) * 256;
    pack_w_kernel<<<(tot + 255) / 256, 256, 0, stream>>>(W3, e3, W3p, 1024, 100, 128);
  }

  gemm_i8_kernel<3072, 1024, true, false, true>
      <<<dim3(8, 64), 256, 0, stream>>>(A1p, W1p, b1, eb1, hiddens, nullptr,
                                        (uint8_t*)A2p);
  gemm_i8_kernel<1024, 1024, false, true, true>
      <<<dim3(8, 64), 256, 0, stream>>>(A2p, W2p, b2, eb2, nullptr, outH2,
                                        (uint8_t*)A3p);
  gemm_i8_kernel<1024, 100, false, true, false>
      <<<dim3(1, 64), 256, 0, stream>>>(A3p, W3p, b3, eb3, nullptr, outO,
                                        nullptr);
}
